// SCAN_84061099917852
// MI455X (gfx1250) — matrix operand order repaired and checked by reading, not compiled, not run on hardware
//
#include <hip/hip_runtime.h>
#include <hip/hip_bf16.h>
#include <math.h>

typedef __attribute__((ext_vector_type(16))) __bf16 v16bf;
typedef __attribute__((ext_vector_type(8)))  float  v8f;
typedef __attribute__((ext_vector_type(4)))  float  v4f;

#define NB 256
#define NR 150
#define ND 1024
#define NW 40
#define NWP 48        // words padded to 3 N-tiles of 16
#define KP  64        // padded K for the attn@G WMMA
#define LAM 6.0f
#define EPSV 1e-6f
#define ES_STRIDE 160
#define DSTR 52       // LDS row stride (floats) for dots/attn

// ---------------- prep kernels (tiny, batch-independent) ----------------

__global__ void k_prep_desc(const float* __restrict__ desc, __bf16* __restrict__ descB) {
    int idx = blockIdx.x * blockDim.x + threadIdx.x;      // 48*1024 threads
    int w = idx >> 10, k = idx & 1023;
    float v = (w < NW) ? desc[w * ND + k] : 0.0f;
    descB[idx] = (__bf16)v;
}

__global__ void k_word_norm(const float* __restrict__ desc, float* __restrict__ wnorm) {
    int w = blockIdx.x;                                   // 48 blocks x 32
    int lane = threadIdx.x;
    float s = 0.f;
    if (w < NW)
        for (int k = lane; k < ND; k += 32) { float x = desc[w * ND + k]; s += x * x; }
    for (int m = 16; m >= 1; m >>= 1) s += __shfl_xor(s, m, 32);
    if (lane == 0) wnorm[w] = (w < NW) ? sqrtf(s) : 1.0f;
}

// Gcol[n][k] = desc_n . desc_k (symmetric), bf16, 48x64 zero-padded.
// One wave per output element; float4 lane-strided loads + butterfly reduce.
__global__ void k_gram(const float* __restrict__ desc, __bf16* __restrict__ G) {
    int n = blockIdx.x >> 6, k = blockIdx.x & 63;         // 48*64 blocks x 32
    int lane = threadIdx.x;
    float s = 0.f;
    if (n < NW && k < NW) {
        const float* a = desc + (size_t)n * ND;
        const float* b = desc + (size_t)k * ND;
        for (int t = lane * 4; t < ND; t += 128) {
            v4f xa = *(const v4f*)(a + t);
            v4f xb = *(const v4f*)(b + t);
            s += xa[0]*xb[0] + xa[1]*xb[1] + xa[2]*xb[2] + xa[3]*xb[3];
        }
    }
    for (int m = 16; m >= 1; m >>= 1) s += __shfl_xor(s, m, 32);
    if (lane == 0) G[blockIdx.x] = (__bf16)s;
}

// ---------------- main kernel: one batch per block, one 16-row tile per wave ----

__global__ void __launch_bounds__(320)
k_main(const float* __restrict__ vis, const __bf16* __restrict__ descB,
       const __bf16* __restrict__ G, const float* __restrict__ wnorm,
       float* __restrict__ expS) {
    __shared__ float  s_dots[10][16 * DSTR];   // dots, later reused as attn(f32)
    __shared__ __bf16 s_attnB[10][16 * KP];    // attn bf16, K-padded to 64
    __shared__ float  s_sent2[10][16];
    __shared__ float  s_wn[NWP];

    const int wv = threadIdx.x >> 5;   // wave id = row tile 0..9
    const int l  = threadIdx.x & 31;
    const int b  = blockIdx.x;
    const int lr = l & 15;             // A-layout: M = lr
    const int hh = l >> 4;             // lane half selects K sub-offset

    if (threadIdx.x < NWP) s_wn[threadIdx.x] = wnorm[threadIdx.x];

    // ---- phase 1: dots(16x48) = vis_tile(16x1024) @ descT via bf16 WMMA ----
    const int row  = wv * 16 + lr;
    const int rowc = (row < NR) ? row : (NR - 1);          // clamp padded rows
    const float* arow = vis + ((size_t)b * NR + rowc) * ND + hh * 16;

    v8f acc0 = {}; v8f acc1 = {}; v8f acc2 = {};
    v4f n0 = {}; v4f n1 = {};                  // ||row||^2 partials (pk_fma-friendly)

    for (int ck = 0; ck < 32; ++ck) {
        const float* p = arow + ck * 32;
        v4f x0 = *(const v4f*)(p);
        v4f x1 = *(const v4f*)(p + 4);
        v4f x2 = *(const v4f*)(p + 8);
        v4f x3 = *(const v4f*)(p + 12);
        n0 = x0 * x0 + n0;  n1 = x1 * x1 + n1;
        n0 = x2 * x2 + n0;  n1 = x3 * x3 + n1;
        v16bf a;
        a[0]=(__bf16)x0[0]; a[1]=(__bf16)x0[1]; a[2]=(__bf16)x0[2]; a[3]=(__bf16)x0[3];
        a[4]=(__bf16)x1[0]; a[5]=(__bf16)x1[1]; a[6]=(__bf16)x1[2]; a[7]=(__bf16)x1[3];
        a[8]=(__bf16)x2[0]; a[9]=(__bf16)x2[1]; a[10]=(__bf16)x2[2]; a[11]=(__bf16)x2[3];
        a[12]=(__bf16)x3[0]; a[13]=(__bf16)x3[1]; a[14]=(__bf16)x3[2]; a[15]=(__bf16)x3[3];

        const __bf16* bp = descB + (size_t)lr * ND + ck * 32 + hh * 16;
        v16bf b0 = *(const v16bf*)(bp);
        v16bf b1 = *(const v16bf*)(bp + (size_t)16 * ND);
        v16bf b2 = *(const v16bf*)(bp + (size_t)32 * ND);
        acc0 = __builtin_amdgcn_wmma_f32_16x16x32_bf16(false, a, false, b0, (short)0, acc0, false, false);
        acc1 = __builtin_amdgcn_wmma_f32_16x16x32_bf16(false, a, false, b1, (short)0, acc1, false, false);
        acc2 = __builtin_amdgcn_wmma_f32_16x16x32_bf16(false, a, false, b2, (short)0, acc2, false, false);
    }
    float nsq = (n0[0] + n0[1]) + (n0[2] + n0[3]) + ((n1[0] + n1[1]) + (n1[2] + n1[3]));
    const float vn = sqrtf(nsq + __shfl_xor(nsq, 16, 32));  // ||vis_row||

    // C/D layout: vgpr i holds M=i (lanes 0-15) / M=8+i (lanes 16-31), N=lane&15
    float* dl = s_dots[wv];
    const int mrow = hh * 8;
    #pragma unroll
    for (int i = 0; i < 8; ++i) {
        dl[(mrow + i) * DSTR + lr]      = acc0[i];
        dl[(mrow + i) * DSTR + 16 + lr] = acc1[i];
        dl[(mrow + i) * DSTR + 32 + lr] = acc2[i];
    }
    __syncthreads();

    // ---- phase 2: softmax over 40 words, 2 lanes per row (lane pair lr / lr+16) ----
    float d[20], e[20];
    float* drow = s_dots[wv] + lr * DSTR;
    float mx = -1e30f;
    #pragma unroll
    for (int t = 0; t < 20; ++t) {
        int w = hh * 20 + t;
        d[t] = drow[w];
        float sc = d[t] / fmaxf(vn * s_wn[w], EPSV);
        e[t] = sc;
        mx = fmaxf(mx, sc);
    }
    mx = fmaxf(mx, __shfl_xor(mx, 16, 32));
    float Z = 0.f, ED = 0.f;
    #pragma unroll
    for (int t = 0; t < 20; ++t) {
        float ee = __expf(e[t] - mx);
        e[t] = ee; Z += ee; ED += ee * d[t];
    }
    Z  += __shfl_xor(Z, 16, 32);
    ED += __shfl_xor(ED, 16, 32);
    const float rZ = 1.0f / Z;
    const float dots2 = ED * rZ;

    __bf16* ab = s_attnB[wv] + lr * KP;
    #pragma unroll
    for (int t = 0; t < 20; ++t) {
        int w = hh * 20 + t;
        float aw = e[t] * rZ;
        drow[w] = aw;                 // overlay f32 attn onto dots storage
        ab[w]   = (__bf16)aw;
    }
    if (hh == 1) {                    // zero pads (w = 40..63 / 40..47)
        for (int t = 0; t < 24; ++t) ab[40 + t] = (__bf16)0.0f;
        for (int t = 0; t < 8;  ++t) drow[40 + t] = 0.0f;
    }
    __syncthreads();

    // ---- phase 3: Y(16x48) = attn(16x64,bf16) @ G(64x48,bf16) via WMMA ----
    v8f y0 = {}; v8f y1 = {}; v8f y2 = {};
    #pragma unroll
    for (int ck = 0; ck < 2; ++ck) {
        v16bf a;
        const __bf16* ap = s_attnB[wv] + lr * KP + ck * 32 + hh * 16;
        #pragma unroll
        for (int q = 0; q < 8; ++q) { a[q] = ap[q]; a[8 + q] = ap[8 + q]; }
        const __bf16* gp = G + (size_t)lr * KP + ck * 32 + hh * 16;
        v16bf g0 = *(const v16bf*)(gp);
        v16bf g1 = *(const v16bf*)(gp + 16 * KP);
        v16bf g2 = *(const v16bf*)(gp + 32 * KP);
        y0 = __builtin_amdgcn_wmma_f32_16x16x32_bf16(false, a, false, g0, (short)0, y0, false, false);
        y1 = __builtin_amdgcn_wmma_f32_16x16x32_bf16(false, a, false, g1, (short)0, y1, false, false);
        y2 = __builtin_amdgcn_wmma_f32_16x16x32_bf16(false, a, false, g2, (short)0, y2, false, false);
    }

    // sent2[m] = sum_n attn[m][n] * Y[m][n]  (butterfly over the 16 lanes of each half)
    float pr[8];
    const float* af = s_dots[wv];
    #pragma unroll
    for (int i = 0; i < 8; ++i) {
        const float* ar = af + (i + mrow) * DSTR;
        pr[i] = y0[i] * ar[lr] + y1[i] * ar[16 + lr] + y2[i] * ar[32 + lr];
    }
    #pragma unroll
    for (int msk = 1; msk < 16; msk <<= 1)
        #pragma unroll
        for (int i = 0; i < 8; ++i) pr[i] += __shfl_xor(pr[i], msk, 32);
    if (lr == 0)
        #pragma unroll
        for (int i = 0; i < 8; ++i) s_sent2[wv][mrow + i] = pr[i];
    __syncthreads();

    const float s2 = s_sent2[wv][lr];
    const float score = dots2 / fmaxf(vn * sqrtf(fmaxf(s2, 0.f)), EPSV);
    const float es = __expf(LAM * score);
    if (hh == 0 && row < NR) expS[(size_t)b * ES_STRIDE + row] = es;
}

// ---------------- epilogue: 21 segment LSEs from 6 block sums of 25 rows -------

__global__ void k_lse(const float* __restrict__ expS, float* __restrict__ out) {
    int b = blockIdx.x, lane = threadIdx.x;        // 256 x 32
    float S[6];
    #pragma unroll
    for (int k = 0; k < 6; ++k) {
        float pv = (lane < 25) ? expS[(size_t)b * ES_STRIDE + k * 25 + lane] : 0.f;
        for (int m = 16; m >= 1; m >>= 1) pv += __shfl_xor(pv, m, 32);
        S[k] = pv;
    }
    if (lane < 21) {
        int c = lane, i = 0;
        while (c >= 6 - i) { c -= 6 - i; ++i; }
        int j = i + c;
        float s = 0.f;
        for (int k = i; k <= j; ++k) s += S[k];
        out[b * 21 + lane] = logf(s) / LAM;
    }
}

extern "C" void kernel_launch(void* const* d_in, const int* in_sizes, int n_in,
                              void* d_out, int out_size, void* d_ws, size_t ws_size,
                              hipStream_t stream) {
    const float* vis  = (const float*)d_in[0];   // (256,150,1024) f32
    const float* desc = (const float*)d_in[1];   // (1,40,1024) f32
    char* ws = (char*)d_ws;
    __bf16* descB = (__bf16*)(ws);               // 48*1024*2  = 98304 B
    __bf16* G     = (__bf16*)(ws + 98304);       // 48*64*2    =  6144 B
    float*  wnorm = (float*)(ws + 104448);       // 48*4       =   192 B
    float*  expS  = (float*)(ws + 104704);       // 256*160*4  = 163840 B

    k_prep_desc<<<dim3(192),  dim3(256), 0, stream>>>(desc, descB);
    k_word_norm<<<dim3(48),   dim3(32),  0, stream>>>(desc, wnorm);
    k_gram     <<<dim3(3072), dim3(32),  0, stream>>>(desc, G);
    k_main     <<<dim3(256),  dim3(320), 0, stream>>>(vis, descB, G, wnorm, expS);
    k_lse      <<<dim3(256),  dim3(32),  0, stream>>>(expS, (float*)d_out);
}
